// GNNClassifier_60524679135240
// MI455X (gfx1250) — compile-verified, matrix operand order checked by reading, not run on hardware
//
#include <hip/hip_runtime.h>
#include <hip/hip_bf16.h>

// ---------------------------------------------------------------------------
// GATv2 GNN (3 layers) + mean-pool + MLP for MI455X (gfx1250, wave32).
// GEMMs use v_wmma_f32_16x16x32_f16 (f16 in, f32 accumulate).
// Edge phase: wave-per-edge coalesced gathers, L2-resident float atomics.
// ---------------------------------------------------------------------------

typedef __attribute__((ext_vector_type(16))) _Float16 v16h;
typedef __attribute__((ext_vector_type(8)))  float    v8f;

#define N_NODES 50000
#define N_EDGES 800000
#define N_ETOT  850000   // + self loops
#define N_GRAPHS 64
#define NEG_SLOPE 0.2f

// ----- helpers --------------------------------------------------------------

__device__ __forceinline__ unsigned f2ord(float f) {
    unsigned u = __float_as_uint(f);
    return (u & 0x80000000u) ? ~u : (u | 0x80000000u);
}
__device__ __forceinline__ float ord2f(unsigned u) {
    return (u & 0x80000000u) ? __uint_as_float(u & 0x7FFFFFFFu)
                             : __uint_as_float(~u);
}

__global__ void zero_u32(unsigned* __restrict__ p, int n) {
    int i = blockIdx.x * blockDim.x + threadIdx.x;
    if (i < n) p[i] = 0u;
}

__global__ void f32_to_f16(const float* __restrict__ src,
                           _Float16* __restrict__ dst, int n) {
    int i = blockIdx.x * blockDim.x + threadIdx.x;
    if (i < n) dst[i] = (_Float16)src[i];
}

// W[K][Nout] (row-major) -> Wt[Nout][K] fp16 (so B-fragment loads mirror A's)
__global__ void transpose_to_f16(const float* __restrict__ W,
                                 _Float16* __restrict__ Wt, int K, int Nout) {
    int i = blockIdx.x * blockDim.x + threadIdx.x;
    if (i >= K * Nout) return;
    int k = i / Nout, n = i - k * Nout;
    Wt[(size_t)n * K + k] = (_Float16)W[i];
}

// ----- WMMA GEMM: out[M][Nout] = A[M][K](f16) * Bt[Nout][K](f16)^T + bias ---
// One wave -> one 16x16 tile. Block = 4 waves covering 64 output columns.
// M multiple of 16, Nout multiple of 64, K multiple of 32 (true for all calls).

__global__ __launch_bounds__(128) void gemm_wmma_bias(
    const _Float16* __restrict__ A, const _Float16* __restrict__ Bt,
    const float* __restrict__ bias, float* __restrict__ out,
    int M, int K, int Nout) {
    const int wave = threadIdx.x >> 5;
    const int lane = threadIdx.x & 31;
    const int m0 = blockIdx.x * 16;
    const int n0 = (blockIdx.y * 4 + wave) * 16;
    if (n0 >= Nout) return;                 // wave-uniform

    const int khalf = lane >> 4;            // 0: k 0-7 and 16-23, 1: k 8-15 and 24-31
    const int mrow  = m0 + (lane & 15);
    const int ncol  = n0 + (lane & 15);
    const _Float16* __restrict__ arow = A  + (size_t)mrow * K;
    const _Float16* __restrict__ brow = Bt + (size_t)ncol * K;

    v8f c = {};
    for (int k0 = 0; k0 < K; k0 += 32) {
        union { v16h v; unsigned u[8]; } a, b;
#pragma unroll
        for (int j = 0; j < 8; ++j) {
            // ISA 7.12.2: 16-bit A 16x32 fragment K index per (VGPR j, lane half)
            const int kb = (j < 4) ? (khalf * 8 + 2 * j)
                                   : (16 + khalf * 8 + 2 * (j - 4));
            a.u[j] = *(const unsigned*)(arow + k0 + kb);
            b.u[j] = *(const unsigned*)(brow + k0 + kb);
        }
        c = __builtin_amdgcn_wmma_f32_16x16x32_f16(
                false, a.v, false, b.v, (short)0, c, false, false);
    }

    const int n  = n0 + (lane & 15);
    const float bv = bias[n];
    const int mb = m0 + (lane >> 4) * 8;
#pragma unroll
    for (int r = 0; r < 8; ++r)
        out[(size_t)(mb + r) * Nout + n] = c[r] + bv;
}

// ----- edge kernels ---------------------------------------------------------

__device__ __forceinline__ void edge_sd(const int* __restrict__ ei, int e,
                                        int& s, int& d) {
    if (e < N_EDGES) { s = ei[e]; d = ei[N_EDGES + e]; }
    else             { s = d = e - N_EDGES; }
}

// logits[e][h] = att[h] . leaky_relu(xl[s] + xr[d]);  segmax via ordered-uint
__global__ void edge_logits(const float* __restrict__ xl,
                            const float* __restrict__ xr,
                            const float* __restrict__ att,
                            const int* __restrict__ ei, int H,
                            float* __restrict__ elog,
                            unsigned* __restrict__ segmax) {
    const int e = blockIdx.x * (blockDim.x >> 5) + (threadIdx.x >> 5);
    if (e >= N_ETOT) return;
    const int lane = threadIdx.x & 31;
    int s, d; edge_sd(ei, e, s, d);
    const int HC = H * 64;
    const float* xls = xl + (size_t)s * HC;
    const float* xrd = xr + (size_t)d * HC;
    for (int h = 0; h < H; ++h) {
        float acc = 0.f;
#pragma unroll
        for (int cc = 0; cc < 2; ++cc) {
            const int c = h * 64 + lane + cc * 32;
            float m = xls[c] + xrd[c];
            m = (m > 0.f) ? m : m * NEG_SLOPE;
            acc += m * att[h * 64 + lane + cc * 32];
        }
#pragma unroll
        for (int off = 16; off > 0; off >>= 1)
            acc += __shfl_xor(acc, off, 32);
        if (lane == 0) {
            elog[(size_t)e * H + h] = acc;
            atomicMax(&segmax[(size_t)d * H + h], f2ord(acc));
        }
    }
}

// elog <- exp(logit - segmax[dst]);  denom[dst] += elog
__global__ void edge_expsum(float* __restrict__ elog,
                            const unsigned* __restrict__ segmax,
                            float* __restrict__ denom,
                            const int* __restrict__ ei,
                            int H, int logH, int total) {
    const int idx = blockIdx.x * blockDim.x + threadIdx.x;
    if (idx >= total) return;
    const int e = idx >> logH;
    const int h = idx & (H - 1);
    const int d = (e < N_EDGES) ? ei[N_EDGES + e] : (e - N_EDGES);
    const float mx = ord2f(segmax[(size_t)d * H + h]);
    const float ex = __expf(elog[idx] - mx);
    elog[idx] = ex;
    atomicAdd(&denom[(size_t)d * H + h], ex);
}

// outacc[dst] += (ex/denom[dst]) * xl[src]     (wave per edge, lanes over ch)
__global__ void edge_scatter(const float* __restrict__ xl,
                             const float* __restrict__ elog,
                             const float* __restrict__ denom,
                             const int* __restrict__ ei, int H,
                             float* __restrict__ outacc) {
    const int e = blockIdx.x * (blockDim.x >> 5) + (threadIdx.x >> 5);
    if (e >= N_ETOT) return;
    const int lane = threadIdx.x & 31;
    int s, d; edge_sd(ei, e, s, d);
    const int HC = H * 64;
    const float* xls = xl + (size_t)s * HC;
    float* od = outacc + (size_t)d * HC;
    for (int h = 0; h < H; ++h) {
        const float alpha = elog[(size_t)e * H + h] / denom[(size_t)d * H + h];
#pragma unroll
        for (int cc = 0; cc < 2; ++cc) {
            const int c = h * 64 + lane + cc * 32;
            atomicAdd(&od[c], alpha * xls[c]);
        }
    }
}

__global__ void finalize_relu_bias(float* __restrict__ h,
                                   const float* __restrict__ b,
                                   int mask, int total) {
    const int i = blockIdx.x * blockDim.x + threadIdx.x;
    if (i >= total) return;
    const float v = h[i] + b[i & mask];
    h[i] = (v > 0.f) ? v : 0.f;
}

// ----- pooling + MLP --------------------------------------------------------

__global__ void pool_sum(const float* __restrict__ h,
                         const int* __restrict__ batch,
                         float* __restrict__ pooled) {
    const int idx = blockIdx.x * blockDim.x + threadIdx.x;
    if (idx >= N_NODES * 64) return;
    const int n = idx >> 6, c = idx & 63;
    atomicAdd(&pooled[batch[n] * 64 + c], h[idx]);
}

__global__ void pool_count(const int* __restrict__ batch,
                           float* __restrict__ counts) {
    const int n = blockIdx.x * blockDim.x + threadIdx.x;
    if (n < N_NODES) atomicAdd(&counts[batch[n]], 1.0f);
}

__global__ void final_mlp(const float* __restrict__ pooled,
                          const float* __restrict__ counts,
                          const float* __restrict__ Wm1,
                          const float* __restrict__ bm1,
                          const float* __restrict__ Wm2,
                          const float* __restrict__ bm2,
                          float* __restrict__ out) {
    const int g = threadIdx.x;
    if (g >= N_GRAPHS) return;
    const float inv = 1.0f / fmaxf(counts[g], 1.0f);
    float mean[64];
#pragma unroll
    for (int c = 0; c < 64; ++c) mean[c] = pooled[g * 64 + c] * inv;
    float hid[32];
    for (int j = 0; j < 32; ++j) {
        float a = bm1[j];
        for (int c = 0; c < 64; ++c) a += mean[c] * Wm1[c * 32 + j];
        hid[j] = fmaxf(a, 0.f);
    }
    for (int t = 0; t < 2; ++t) {
        float a = bm2[t];
        for (int j = 0; j < 32; ++j) a += hid[j] * Wm2[j * 2 + t];
        out[g * 2 + t] = a;
    }
}

// ----- host driver ----------------------------------------------------------

extern "C" void kernel_launch(void* const* d_in, const int* in_sizes, int n_in,
                              void* d_out, int out_size, void* d_ws, size_t ws_size,
                              hipStream_t stream) {
    const float* x   = (const float*)d_in[0];
    const int* ei    = (const int*)d_in[1];
    const int* batch = (const int*)d_in[2];
    const float *Wl1=(const float*)d_in[3],  *bl1=(const float*)d_in[4];
    const float *Wr1=(const float*)d_in[5],  *br1=(const float*)d_in[6];
    const float *att1=(const float*)d_in[7], *b1=(const float*)d_in[8];
    const float *Wl2=(const float*)d_in[9],  *bl2=(const float*)d_in[10];
    const float *Wr2=(const float*)d_in[11], *br2=(const float*)d_in[12];
    const float *att2=(const float*)d_in[13],*b2=(const float*)d_in[14];
    const float *Wl3=(const float*)d_in[15], *bl3=(const float*)d_in[16];
    const float *Wr3=(const float*)d_in[17], *br3=(const float*)d_in[18];
    const float *att3=(const float*)d_in[19],*b3=(const float*)d_in[20];
    const float *Wm1=(const float*)d_in[21], *bm1=(const float*)d_in[22];
    const float *Wm2=(const float*)d_in[23], *bm2=(const float*)d_in[24];

    const int N = N_NODES, ET = N_ETOT;

    char* ws = (char*)d_ws;
    size_t off = 0;
    auto take = [&](size_t bytes) -> void* {
        void* p = ws + off;
        off = (off + bytes + 255) & ~(size_t)255;
        return p;
    };
    _Float16* xh   = (_Float16*)take((size_t)N * 256 * 2);
    _Float16* wlh  = (_Float16*)take(1 << 17);
    _Float16* wrh  = (_Float16*)take(1 << 17);
    float* xl      = (float*)take((size_t)N * 256 * 4);
    float* xr      = (float*)take((size_t)N * 256 * 4);
    float* hbuf    = (float*)take((size_t)N * 256 * 4);
    float* elog    = (float*)take((size_t)ET * 4 * 4);
    unsigned* smax = (unsigned*)take((size_t)N * 4 * 4);
    float* denom   = (float*)take((size_t)N * 4 * 4);
    float* pooled  = (float*)take(N_GRAPHS * 64 * 4);
    float* counts  = (float*)take(N_GRAPHS * 4);
    (void)ws_size; (void)n_in; (void)in_sizes; (void)out_size;

    const int EB = (ET + 7) / 8;  // blocks of 8 waves, wave-per-edge

    auto run_layer = [&](const float* hin, int K, int H,
                         const float* Wl, const float* bl,
                         const float* Wr, const float* br,
                         const float* att, const float* bout) {
        const int HC = H * 64;
        const int nK = N * K;
        f32_to_f16<<<(nK + 255) / 256, 256, 0, stream>>>(hin, xh, nK);
        transpose_to_f16<<<(K * HC + 255) / 256, 256, 0, stream>>>(Wl, wlh, K, HC);
        transpose_to_f16<<<(K * HC + 255) / 256, 256, 0, stream>>>(Wr, wrh, K, HC);
        dim3 gg(N / 16, HC / 64);
        gemm_wmma_bias<<<gg, 128, 0, stream>>>(xh, wlh, bl, xl, N, K, HC);
        gemm_wmma_bias<<<gg, 128, 0, stream>>>(xh, wrh, br, xr, N, K, HC);
        zero_u32<<<(N * H + 255) / 256, 256, 0, stream>>>(smax, N * H);
        zero_u32<<<(N * H + 255) / 256, 256, 0, stream>>>((unsigned*)denom, N * H);
        zero_u32<<<(N * HC + 255) / 256, 256, 0, stream>>>((unsigned*)hbuf, N * HC);
        edge_logits<<<EB, 256, 0, stream>>>(xl, xr, att, ei, H, elog, smax);
        const int logH = (H == 4) ? 2 : (H == 2) ? 1 : 0;
        const int tot = ET * H;
        edge_expsum<<<(tot + 255) / 256, 256, 0, stream>>>(elog, smax, denom, ei,
                                                           H, logH, tot);
        edge_scatter<<<EB, 256, 0, stream>>>(xl, elog, denom, ei, H, hbuf);
        finalize_relu_bias<<<(N * HC + 255) / 256, 256, 0, stream>>>(
            hbuf, bout, HC - 1, N * HC);
    };

    // conv1: in=128, H=4, C=64, concat -> N x 256
    run_layer(x,    128, 4, Wl1, bl1, Wr1, br1, att1, b1);
    // conv2: in=256, H=2, C=64, concat -> N x 128
    run_layer(hbuf, 256, 2, Wl2, bl2, Wr2, br2, att2, b2);
    // conv3: in=128, H=1, C=64, mean over 1 head -> N x 64
    run_layer(hbuf, 128, 1, Wl3, bl3, Wr3, br3, att3, b3);

    // global mean pool + MLP
    zero_u32<<<(N_GRAPHS * 64 + 255) / 256, 256, 0, stream>>>((unsigned*)pooled,
                                                              N_GRAPHS * 64);
    zero_u32<<<1, 64, 0, stream>>>((unsigned*)counts, N_GRAPHS);
    pool_sum<<<(N * 64 + 255) / 256, 256, 0, stream>>>(hbuf, batch, pooled);
    pool_count<<<(N + 255) / 256, 256, 0, stream>>>(batch, counts);
    final_mlp<<<1, 64, 0, stream>>>(pooled, counts, Wm1, bm1, Wm2, bm2,
                                    (float*)d_out);
}
